// Memory_17205638987962
// MI455X (gfx1250) — matrix operand order repaired and checked by reading, not compiled, not run on hardware
//
#include <hip/hip_runtime.h>
#include <hip/hip_bf16.h>

typedef __attribute__((ext_vector_type(16))) __bf16 v16bf;
typedef __attribute__((ext_vector_type(8)))  __bf16 v8bf;
typedef __attribute__((ext_vector_type(8)))  float  v8f;

static constexpr int B_ = 4, DK = 128, DV = 512, LM = 4096, LQ = 4096;

// ---------------------------------------------------------------------------
// Kernel 1: E[b,m,q] = exp( (1/sqrt(DK)) * sum_d mk[b,d,m]*qk[b,d,q] )
// Tile 128(m) x 128(q), K=128 fully resident in LDS as bf16.
// 8 waves = 2(m) x 4(q); each wave does 4x2 tiles of v_wmma_f32_16x16x32_bf16.
// ---------------------------------------------------------------------------
__global__ void __launch_bounds__(256)
k_scores(const float* __restrict__ mk, const float* __restrict__ qk,
         float* __restrict__ p_out)
{
    constexpr int BM = 128, BN = 128, KD = 128, LDK = KD + 8; // pad 8 bf16
    __shared__ __align__(16) __bf16 lA[BM * LDK];   // [m][k]
    __shared__ __align__(16) __bf16 lB[BN * LDK];   // [q][k]

    const int b  = blockIdx.y;
    const int tm = (blockIdx.x & 31) * BM;
    const int tn = (blockIdx.x >> 5) * BN;
    const int t  = threadIdx.x;

    const float* gA = mk + (size_t)b * DK * LM;   // [k][m], m contiguous
    const float* gB = qk + (size_t)b * DK * LQ;   // [k][q], q contiguous

    // Stage + transpose + cvt f32->bf16: 128 k-rows x 32 float4 each.
    for (int v = t; v < KD * (BM / 4); v += 256) {
        const int k  = v >> 5;
        const int x4 = (v & 31) * 4;
        float4 a = *(const float4*)(gA + (size_t)k * LM + tm + x4);
        lA[(x4 + 0) * LDK + k] = (__bf16)a.x;
        lA[(x4 + 1) * LDK + k] = (__bf16)a.y;
        lA[(x4 + 2) * LDK + k] = (__bf16)a.z;
        lA[(x4 + 3) * LDK + k] = (__bf16)a.w;
        float4 c = *(const float4*)(gB + (size_t)k * LQ + tn + x4);
        lB[(x4 + 0) * LDK + k] = (__bf16)c.x;
        lB[(x4 + 1) * LDK + k] = (__bf16)c.y;
        lB[(x4 + 2) * LDK + k] = (__bf16)c.z;
        lB[(x4 + 3) * LDK + k] = (__bf16)c.w;
    }
    __syncthreads();

    const int lane  = t & 31;
    const int lmod  = lane & 15;
    const int lhalf = lane >> 4;
    const int wave  = t >> 5;
    const int wm    = (wave >> 2) * 64;   // 0,64
    const int wn    = (wave & 3) * 32;    // 0..96

    v8f acc[4][2] = {};

    #pragma unroll
    for (int k0 = 0; k0 < KD; k0 += 32) {
        v16bf af[4], bf[2];
        #pragma unroll
        for (int mt = 0; mt < 4; ++mt) {
            // A 16x32 bf16: lanes0-15 K=0..7 (v0-3) & 16..23 (v4-7); lanes16-31 +8
            const __bf16* pa = &lA[(wm + mt * 16 + lmod) * LDK + k0 + lhalf * 8];
            v8bf lo = *(const v8bf*)pa;
            v8bf hi = *(const v8bf*)(pa + 16);
            af[mt] = __builtin_shufflevector(lo, hi, 0,1,2,3,4,5,6,7,8,9,10,11,12,13,14,15);
        }
        #pragma unroll
        for (int nt = 0; nt < 2; ++nt) {
            // B 32x16 bf16: lanes0-15 K=0..7 & 16..23; lanes16-31 +8
            const __bf16* pb = &lB[(wn + nt * 16 + lmod) * LDK + k0 + lhalf * 8];
            v8bf lo = *(const v8bf*)pb;
            v8bf hi = *(const v8bf*)(pb + 16);
            bf[nt] = __builtin_shufflevector(lo, hi, 0,1,2,3,4,5,6,7,8,9,10,11,12,13,14,15);
        }
        #pragma unroll
        for (int mt = 0; mt < 4; ++mt)
            #pragma unroll
            for (int nt = 0; nt < 2; ++nt)
                acc[mt][nt] = __builtin_amdgcn_wmma_f32_16x16x32_bf16(
                    false, af[mt], false, bf[nt], (short)0, acc[mt][nt], false, false);
    }

    const float scale = 0.0883883476483184405f; // 1/sqrt(128)
    float* prow = p_out + (size_t)b * LM * LQ;
    #pragma unroll
    for (int mt = 0; mt < 4; ++mt)
        #pragma unroll
        for (int nt = 0; nt < 2; ++nt) {
            const int q = tn + wn + nt * 16 + lmod;
            #pragma unroll
            for (int r = 0; r < 8; ++r) {
                const int m = tm + wm + mt * 16 + lhalf * 8 + r;
                prow[(size_t)m * LQ + q] = __expf(acc[mt][nt][r] * scale);
            }
        }
}

// ---------------------------------------------------------------------------
// Kernel 2: in-place column softmax normalization: p[:,q] /= sum_m p[m,q].
// Block = 64 columns of one batch; 4 threads per column; 4 independent
// accumulators per thread (fixed order -> deterministic).
// ---------------------------------------------------------------------------
__global__ void __launch_bounds__(256)
k_norm(float* __restrict__ p)
{
    __shared__ float sred[256];
    const int blk = blockIdx.x;               // B * (LQ/64)
    const int b   = blk / (LQ / 64);
    const int q0  = (blk % (LQ / 64)) * 64;
    const int t   = threadIdx.x;
    const int ql  = t & 63;
    const int ms  = t >> 6;                   // 0..3 row slice
    float* base = p + (size_t)b * LM * LQ + q0 + ql;

    float s0 = 0.f, s1 = 0.f, s2 = 0.f, s3 = 0.f;
    for (int m = ms; m < LM; m += 16) {
        s0 += base[(size_t)(m     ) * LQ];
        s1 += base[(size_t)(m +  4) * LQ];
        s2 += base[(size_t)(m +  8) * LQ];
        s3 += base[(size_t)(m + 12) * LQ];
    }
    sred[t] = (s0 + s1) + (s2 + s3);
    __syncthreads();
    const float inv = 1.0f / (sred[ql] + sred[ql + 64] + sred[ql + 128] + sred[ql + 192]);
    for (int m = ms; m < LM; m += 4) base[(size_t)m * LQ] *= inv;
}

// ---------------------------------------------------------------------------
// Kernel 3: mem[b,v,q] = sum_m mv[b,v,m] * p[b,m,q].  M=DV, N=LQ, K=LM.
// Tile 256(v) x 128(q), BK=64, 512 threads (16 waves = 4x4).
// Double-buffered LDS: issue global loads for tile k+1, compute tile k,
// then cvt+store tile k+1, one barrier per iteration.
// ---------------------------------------------------------------------------
__global__ void __launch_bounds__(512)
k_out(const float* __restrict__ mv, const float* __restrict__ p,
      float* __restrict__ memo)
{
    constexpr int BM = 256, BN = 128, BK = 64, LDK = BK + 8; // 72
    __shared__ __align__(16) __bf16 lA[2][BM * LDK];   // [v][k]  2 x 36 KB
    __shared__ __align__(16) __bf16 lB[2][BN * LDK];   // [q][k]  2 x 18 KB

    const int b  = blockIdx.y;
    const int tv = (blockIdx.x & 1) * BM;     // 512/256 = 2
    const int tq = (blockIdx.x >> 1) * BN;    // 4096/128 = 32
    const int t  = threadIdx.x;
    const int lane  = t & 31;
    const int lmod  = lane & 15;
    const int lhalf = lane >> 4;
    const int wave  = t >> 5;                 // 0..15
    const int wm    = (wave >> 2) * 64;       // 0..192
    const int wn    = (wave & 3) * 32;        // 0..96

    const float* gA = mv + (size_t)b * DV * LM;   // [v][m], m contiguous
    const float* gB = p  + (size_t)b * LM * LQ;   // [m][q], q contiguous

    float4 aR[8];   // staged A: 256 rows x 16 float4 = 4096 / 512 threads
    float4 bR[4];   // staged B:  64 rows x 32 float4 = 2048 / 512 threads

    auto loadA = [&](int kb) {
        #pragma unroll
        for (int j = 0; j < 8; ++j) {
            const int v = t + j * 512, row = v >> 4, c4 = (v & 15) * 4;
            aR[j] = *(const float4*)(gA + (size_t)(tv + row) * LM + kb + c4);
        }
    };
    auto loadB = [&](int kb) {
        #pragma unroll
        for (int j = 0; j < 4; ++j) {
            const int v = t + j * 512, k = v >> 5, q4 = (v & 31) * 4;
            bR[j] = *(const float4*)(gB + (size_t)(kb + k) * LQ + tq + q4);
        }
    };
    auto storeA = [&](int buf) {
        #pragma unroll
        for (int j = 0; j < 8; ++j) {
            const int v = t + j * 512, row = v >> 4, c4 = (v & 15) * 4;
            __bf16* dst = &lA[buf][row * LDK + c4];
            dst[0] = (__bf16)aR[j].x; dst[1] = (__bf16)aR[j].y;
            dst[2] = (__bf16)aR[j].z; dst[3] = (__bf16)aR[j].w;
        }
    };
    auto storeB = [&](int buf) {    // transpose [m][q] -> [q][k]
        #pragma unroll
        for (int j = 0; j < 4; ++j) {
            const int v = t + j * 512, k = v >> 5, q4 = (v & 31) * 4;
            lB[buf][(q4 + 0) * LDK + k] = (__bf16)bR[j].x;
            lB[buf][(q4 + 1) * LDK + k] = (__bf16)bR[j].y;
            lB[buf][(q4 + 2) * LDK + k] = (__bf16)bR[j].z;
            lB[buf][(q4 + 3) * LDK + k] = (__bf16)bR[j].w;
        }
    };

    v8f acc[4][2] = {};

    loadA(0); loadB(0);
    storeA(0); storeB(0);
    __syncthreads();

    int cur = 0;
    for (int kb = 0; kb < LM; kb += BK) {
        const bool more = (kb + BK) < LM;
        if (more) { loadA(kb + BK); loadB(kb + BK); }
        if (kb + 2 * BK < LM) {  // speculative prefetch two tiles ahead
            __builtin_prefetch(gA + (size_t)(tv + (t & 255)) * LM + kb + 2 * BK, 0, 3);
            __builtin_prefetch(gB + (size_t)(kb + 2 * BK + (t & 63)) * LQ + tq + ((t >> 6) & 1) * 64, 0, 3);
        }

        #pragma unroll
        for (int k0 = 0; k0 < BK; k0 += 32) {
            v16bf af[4], bf[2];
            #pragma unroll
            for (int mt = 0; mt < 4; ++mt) {
                const __bf16* pa = &lA[cur][(wm + mt * 16 + lmod) * LDK + k0 + lhalf * 8];
                v8bf lo = *(const v8bf*)pa;
                v8bf hi = *(const v8bf*)(pa + 16);
                af[mt] = __builtin_shufflevector(lo, hi, 0,1,2,3,4,5,6,7,8,9,10,11,12,13,14,15);
            }
            #pragma unroll
            for (int nt = 0; nt < 2; ++nt) {
                const __bf16* pb = &lB[cur][(wn + nt * 16 + lmod) * LDK + k0 + lhalf * 8];
                v8bf lo = *(const v8bf*)pb;
                v8bf hi = *(const v8bf*)(pb + 16);
                bf[nt] = __builtin_shufflevector(lo, hi, 0,1,2,3,4,5,6,7,8,9,10,11,12,13,14,15);
            }
            #pragma unroll
            for (int mt = 0; mt < 4; ++mt)
                #pragma unroll
                for (int nt = 0; nt < 2; ++nt)
                    acc[mt][nt] = __builtin_amdgcn_wmma_f32_16x16x32_bf16(
                        false, af[mt], false, bf[nt], (short)0, acc[mt][nt], false, false);
        }

        if (more) { storeA(cur ^ 1); storeB(cur ^ 1); }
        __syncthreads();
        cur ^= 1;
    }

    float* out = memo + (size_t)b * DV * LQ;
    #pragma unroll
    for (int mt = 0; mt < 4; ++mt)
        #pragma unroll
        for (int nt = 0; nt < 2; ++nt) {
            const int q = tq + wn + nt * 16 + lmod;
            #pragma unroll
            for (int r = 0; r < 8; ++r) {
                const int v = tv + wm + mt * 16 + lhalf * 8 + r;
                out[(size_t)v * LQ + q] = acc[mt][nt][r];
            }
        }
}

// ---------------------------------------------------------------------------
extern "C" void kernel_launch(void* const* d_in, const int* in_sizes, int n_in,
                              void* d_out, int out_size, void* d_ws, size_t ws_size,
                              hipStream_t stream)
{
    (void)in_sizes; (void)n_in; (void)out_size; (void)d_ws; (void)ws_size;
    const float* m_k = (const float*)d_in[0];
    const float* m_v = (const float*)d_in[1];
    const float* q_k = (const float*)d_in[2];

    float* memo = (float*)d_out;                       // [B, DV, LQ]
    float* p    = memo + (size_t)B_ * DV * LQ;         // [B, LM, LQ]

    // 1) unnormalized exp-scores into p region
    k_scores<<<dim3((LM / 128) * (LQ / 128), B_), dim3(256), 0, stream>>>(m_k, q_k, p);
    // 2) in-place column softmax normalization
    k_norm<<<dim3(B_ * (LQ / 64)), dim3(256), 0, stream>>>(p);
    // 3) mem = mv @ p
    k_out<<<dim3((DV / 256) * (LQ / 128), B_), dim3(512), 0, stream>>>(m_v, p, memo);
}
